// AttentionLayer_35579509080067
// MI455X (gfx1250) — matrix operand order repaired and checked by reading, not compiled, not run on hardware
//
#include <hip/hip_runtime.h>

#define DEV __device__ __forceinline__

typedef __attribute__((ext_vector_type(16))) __bf16 v16bf;
typedef __attribute__((ext_vector_type(8)))  float  v8f;

union AFrag {
  v16bf v;
  uint4 q[2];
};

DEV unsigned short f32_bf16(float f) {
  unsigned int u = __float_as_uint(f);
  u += 0x7FFFu + ((u >> 16) & 1u);   // round-to-nearest-even
  return (unsigned short)(u >> 16);
}

// ---- gfx1250 async global->LDS copies (ASYNCcnt-tracked, bypass VGPRs) ----
DEV void async_ld_b128(void* lds, const void* gaddr) {
  unsigned loff = (unsigned)(uintptr_t)lds;   // generic ptr low 32b == LDS offset
  asm volatile("global_load_async_to_lds_b128 %0, %1, off"
               :: "v"(loff), "v"(gaddr) : "memory");
}
DEV void async_ld_b64(void* lds, const void* gaddr) {
  unsigned loff = (unsigned)(uintptr_t)lds;
  asm volatile("global_load_async_to_lds_b64 %0, %1, off"
               :: "v"(loff), "v"(gaddr) : "memory");
}
DEV void wait_async0() {
  asm volatile("s_wait_asynccnt 0x0" ::: "memory");
}

// ---- pure-VALU 16-lane butterfly reductions (per 16-lane row, wave32) ----
DEV float lane_xor1(float x) {
  return __int_as_float(__builtin_amdgcn_update_dpp(
      __float_as_int(x), __float_as_int(x), 0xB1, 0xF, 0xF, true)); // [1,0,3,2]
}
DEV float lane_xor2(float x) {
  return __int_as_float(__builtin_amdgcn_update_dpp(
      __float_as_int(x), __float_as_int(x), 0x4E, 0xF, 0xF, true)); // [2,3,0,1]
}
DEV float lane_xor4(float x) {
  int v = __float_as_int(x);
  return __int_as_float(__builtin_amdgcn_permlane16(
      v, v, 0x32107654u, 0xBA98FEDCu, false, false));
}
DEV float lane_xor8(float x) {
  int v = __float_as_int(x);
  return __int_as_float(__builtin_amdgcn_permlane16(
      v, v, 0xFEDCBA98u, 0x76543210u, false, false));
}
DEV float red_max16(float x) {
  x = fmaxf(x, lane_xor1(x));
  x = fmaxf(x, lane_xor2(x));
  x = fmaxf(x, lane_xor4(x));
  x = fmaxf(x, lane_xor8(x));
  return x;
}
DEV float red_sum16(float x) {
  x += lane_xor1(x);
  x += lane_xor2(x);
  x += lane_xor4(x);
  x += lane_xor8(x);
  return x;
}

constexpr int D_MODEL = 1024;
constexpr int SEQ     = 2048;
constexpr int BATCH   = 4;
constexpr int HEADS   = 16;
constexpr int HDIM    = 64;
constexpr int MROWS   = BATCH * SEQ;   // 8192

// ---------------------------------------------------------------------------
// GEMM:  Out[M,1024] = A[M,1024] @ W[1024,1024] + bias
//   MODE 0: write bf16 into [B,H,L,E] layout (QKV projections)
//   MODE 1: write fp32 row-major (final output projection)
// Block: 128x64 tile, 256 threads = 8 waves (4x2), each wave 32x32 (2x2 WMMA).
// A staging: fp32 source converts through VGPRs; bf16 source uses async copies.
// ---------------------------------------------------------------------------
template<int MODE, typename AT>
__global__ __launch_bounds__(256) void gemm_bias_k(
    const AT* __restrict__ A, const float* __restrict__ W,
    const float* __restrict__ bias, void* __restrict__ OutP)
{
  __shared__ unsigned short As[128 * 40];   // [m][k], stride 40 halves
  __shared__ unsigned short Bs[64 * 40];    // [n][k] (transposed), stride 40

  const int n0   = blockIdx.x * 64;
  const int m0   = blockIdx.y * 128;
  const int t    = threadIdx.x;
  const int wid  = t >> 5, lane = t & 31;
  const int lm   = lane & 15, lh = lane >> 4;
  const int wm   = wid >> 1, wn = wid & 1;

  v8f acc[2][2];
  #pragma unroll
  for (int a = 0; a < 2; ++a)
    #pragma unroll
    for (int b = 0; b < 2; ++b)
      #pragma unroll
      for (int r = 0; r < 8; ++r) acc[a][b][r] = 0.0f;

  const int ak = (t & 7) * 4;   // A staging: k sub-offset
  const int am = t >> 3;        // A staging: base row (0..31)

  for (int k0 = 0; k0 < D_MODEL; k0 += 32) {
    // ---- stage A panel (128x32) as bf16 ----
    #pragma unroll
    for (int i = 0; i < 4; ++i) {
      const int m = am + 32 * i;
      const AT* ap = A + (size_t)(m0 + m) * D_MODEL + k0 + ak;
      if constexpr (sizeof(AT) == 4) {
        float4 f = *(const float4*)ap;
        unsigned int p0 = (unsigned int)f32_bf16(f.x) | ((unsigned int)f32_bf16(f.y) << 16);
        unsigned int p1 = (unsigned int)f32_bf16(f.z) | ((unsigned int)f32_bf16(f.w) << 16);
        *(uint2*)&As[m * 40 + ak] = make_uint2(p0, p1);
      } else {
        async_ld_b64(&As[m * 40 + ak], ap);   // straight bf16 copy -> async path
      }
    }
    // ---- stage B panel (32k x 64n) transposed to [n][k] bf16 ----
    {
      const int kk = t >> 4;          // 0..15
      const int n4 = (t & 15) * 4;
      #pragma unroll
      for (int i = 0; i < 2; ++i) {
        const int k = kk + 16 * i;
        float4 f = *(const float4*)(W + (size_t)(k0 + k) * D_MODEL + n0 + n4);
        Bs[(n4 + 0) * 40 + k] = f32_bf16(f.x);
        Bs[(n4 + 1) * 40 + k] = f32_bf16(f.y);
        Bs[(n4 + 2) * 40 + k] = f32_bf16(f.z);
        Bs[(n4 + 3) * 40 + k] = f32_bf16(f.w);
      }
    }
    // prefetch next K-panel (global_prefetch_b8)
    if (k0 + 32 < D_MODEL) {
      __builtin_prefetch(A + (size_t)(m0 + am) * D_MODEL + k0 + 32 + ak, 0, 1);
      __builtin_prefetch(W + (size_t)(k0 + 32 + (t >> 4)) * D_MODEL + n0 + (t & 15) * 4, 0, 1);
    }
    if constexpr (sizeof(AT) == 2) wait_async0();
    __syncthreads();

    AFrag af[2], bfr[2];
    #pragma unroll
    for (int tm = 0; tm < 2; ++tm) {
      const int row = wm * 32 + tm * 16 + lm;
      af[tm].q[0] = *(const uint4*)&As[row * 40 + 8 * lh];        // K = 8h..8h+7
      af[tm].q[1] = *(const uint4*)&As[row * 40 + 16 + 8 * lh];   // K = 16+8h..
    }
    #pragma unroll
    for (int tn = 0; tn < 2; ++tn) {
      const int col = wn * 32 + tn * 16 + lm;
      bfr[tn].q[0] = *(const uint4*)&Bs[col * 40 + 8 * lh];       // K = 8h..8h+7
      bfr[tn].q[1] = *(const uint4*)&Bs[col * 40 + 16 + 8 * lh];  // K = 16+8h..
    }
    #pragma unroll
    for (int tm = 0; tm < 2; ++tm)
      #pragma unroll
      for (int tn = 0; tn < 2; ++tn)
        acc[tm][tn] = __builtin_amdgcn_wmma_f32_16x16x32_bf16(
            false, af[tm].v, false, bfr[tn].v, (short)0, acc[tm][tn], false, false);
    __syncthreads();
  }

  // ---- epilogue: bias + store ----
  #pragma unroll
  for (int tm = 0; tm < 2; ++tm) {
    #pragma unroll
    for (int tn = 0; tn < 2; ++tn) {
      const int n  = n0 + wn * 32 + tn * 16 + lm;
      const float bn = bias[n];
      #pragma unroll
      for (int r = 0; r < 8; ++r) {
        const int mg = m0 + wm * 32 + tm * 16 + r + 8 * lh;
        const float vv = acc[tm][tn][r] + bn;
        if constexpr (MODE == 0) {
          const int b = mg >> 11, l = mg & (SEQ - 1);
          const int h = n >> 6,  e = n & (HDIM - 1);
          ((unsigned short*)OutP)[(((size_t)(b * HEADS + h) * SEQ) + l) * HDIM + e] =
              f32_bf16(vv);
        } else {
          ((float*)OutP)[(size_t)mg * D_MODEL + n] = vv;
        }
      }
    }
  }
}

// ---------------------------------------------------------------------------
// Flash attention: one block per (b*h, 128 query rows). 8 waves x 16 rows.
// Streams keys/values in 64-wide chunks; scores never touch HBM.
// K staging uses async global->LDS copies; softmax is pure VALU (DPP/permlane).
// ---------------------------------------------------------------------------
__global__ __launch_bounds__(256) void attn_k(
    const unsigned short* __restrict__ Q, const unsigned short* __restrict__ K,
    const unsigned short* __restrict__ V, unsigned short* __restrict__ O)
{
  __shared__ unsigned short Ks[64 * 72];    // [s][e], stride 72
  __shared__ unsigned short Vt[64 * 72];    // [e][s], stride 72 (transposed)
  __shared__ unsigned short Ps[8][16 * 72]; // per-wave P staging, stride 72

  const int bh  = blockIdx.y;
  const int q0  = blockIdx.x * 128;
  const int t   = threadIdx.x;
  const int wid = t >> 5, lane = t & 31;
  const int lm  = lane & 15, lh = lane >> 4;
  const size_t base_bh = (size_t)bh * SEQ * HDIM;

  // scale = 1/sqrt(64) folded with log2(e) so exponentials are single v_exp_f32
  constexpr float SCL = 0.125f * 1.44269504088896340736f;

  // Loop-invariant Q fragments (k = e dimension, 64 = 2 WMMA steps)
  AFrag aq[2];
  {
    const int qr = q0 + wid * 16 + lm;
    const unsigned short* qp = Q + base_bh + (size_t)qr * HDIM;
    #pragma unroll
    for (int kk = 0; kk < 2; ++kk) {
      aq[kk].q[0] = *(const uint4*)(qp + kk * 32 + 8 * lh);
      aq[kk].q[1] = *(const uint4*)(qp + kk * 32 + 16 + 8 * lh);
    }
  }

  v8f o[4];
  float mi[8], li[8];
  #pragma unroll
  for (int et = 0; et < 4; ++et)
    #pragma unroll
    for (int r = 0; r < 8; ++r) o[et][r] = 0.0f;
  #pragma unroll
  for (int r = 0; r < 8; ++r) { mi[r] = -3.0e38f; li[r] = 0.0f; }

  for (int s0 = 0; s0 < SEQ; s0 += 64) {
    __syncthreads();
    // ---- stage K chunk [64][64] -> Ks[s][e] via async copies ----
    #pragma unroll
    for (int i = 0; i < 2; ++i) {
      const int u = t + 256 * i;
      const int s = u >> 3, e0 = (u & 7) * 8;
      async_ld_b128(&Ks[s * 72 + e0],
                    K + base_bh + (size_t)(s0 + s) * HDIM + e0);
    }
    // ---- stage V chunk transposed -> Vt[e][s] (manual: needs transpose) ----
    #pragma unroll
    for (int i = 0; i < 2; ++i) {
      const int u = t + 256 * i;
      const int s = u & 63, e0 = (u >> 6) * 8;
      union { uint4 q; unsigned short h[8]; } uq;
      uq.q = *(const uint4*)(V + base_bh + (size_t)(s0 + s) * HDIM + e0);
      #pragma unroll
      for (int j = 0; j < 8; ++j) Vt[(e0 + j) * 72 + s] = uq.h[j];
    }
    // prefetch next chunk
    if (s0 + 64 < SEQ) {
      __builtin_prefetch(K + base_bh + (size_t)(s0 + 64 + (t >> 3)) * HDIM + (t & 7) * 8, 0, 1);
      __builtin_prefetch(V + base_bh + (size_t)(s0 + 64 + (t & 63)) * HDIM + (t >> 6) * 16, 0, 1);
    }
    wait_async0();
    __syncthreads();

    // ---- S = Q @ K^T : 4 col tiles x (k=64 -> 2 WMMA) ----
    v8f sacc[4];
    #pragma unroll
    for (int ct = 0; ct < 4; ++ct) {
      const int col = ct * 16 + lm;
      AFrag b0, b1;
      b0.q[0] = *(const uint4*)&Ks[col * 72 + 8 * lh];
      b0.q[1] = *(const uint4*)&Ks[col * 72 + 16 + 8 * lh];
      b1.q[0] = *(const uint4*)&Ks[col * 72 + 32 + 8 * lh];
      b1.q[1] = *(const uint4*)&Ks[col * 72 + 32 + 16 + 8 * lh];
      v8f z;
      #pragma unroll
      for (int r = 0; r < 8; ++r) z[r] = 0.0f;
      z = __builtin_amdgcn_wmma_f32_16x16x32_bf16(
              false, aq[0].v, false, b0.v, (short)0, z, false, false);
      sacc[ct] = __builtin_amdgcn_wmma_f32_16x16x32_bf16(
              false, aq[1].v, false, b1.v, (short)0, z, false, false);
    }

    // ---- online softmax (row = r + 8*lh; its 16 cols live in one 16-lane row) ----
    unsigned short* pw = &Ps[wid][0];
    #pragma unroll
    for (int r = 0; r < 8; ++r) {
      float vv0 = sacc[0][r] * SCL;
      float vv1 = sacc[1][r] * SCL;
      float vv2 = sacc[2][r] * SCL;
      float vv3 = sacc[3][r] * SCL;
      float rmax = red_max16(fmaxf(fmaxf(vv0, vv1), fmaxf(vv2, vv3)));
      const float mnew  = fmaxf(mi[r], rmax);
      const float alpha = __builtin_amdgcn_exp2f(mi[r] - mnew);
      mi[r] = mnew;
      const float e0 = __builtin_amdgcn_exp2f(vv0 - mnew);
      const float e1 = __builtin_amdgcn_exp2f(vv1 - mnew);
      const float e2 = __builtin_amdgcn_exp2f(vv2 - mnew);
      const float e3 = __builtin_amdgcn_exp2f(vv3 - mnew);
      const float rs = red_sum16((e0 + e1) + (e2 + e3));
      li[r] = li[r] * alpha + rs;
      #pragma unroll
      for (int et = 0; et < 4; ++et) o[et][r] *= alpha;
      const int row = r + 8 * lh;             // C-layout -> LDS (A-layout convert)
      pw[row * 72 +  0 + lm] = f32_bf16(e0);
      pw[row * 72 + 16 + lm] = f32_bf16(e1);
      pw[row * 72 + 32 + lm] = f32_bf16(e2);
      pw[row * 72 + 48 + lm] = f32_bf16(e3);
    }

    // P as A-fragments (k = s, 64 -> 2 steps); wave-private LDS round-trip
    AFrag ap[2];
    #pragma unroll
    for (int kk = 0; kk < 2; ++kk) {
      ap[kk].q[0] = *(const uint4*)&pw[lm * 72 + kk * 32 + 8 * lh];
      ap[kk].q[1] = *(const uint4*)&pw[lm * 72 + kk * 32 + 16 + 8 * lh];
    }

    // ---- O += P @ V : 4 col tiles of e x (k=64 -> 2 WMMA) ----
    #pragma unroll
    for (int et = 0; et < 4; ++et) {
      const int col = et * 16 + lm;
      AFrag bv0, bv1;
      bv0.q[0] = *(const uint4*)&Vt[col * 72 + 8 * lh];
      bv0.q[1] = *(const uint4*)&Vt[col * 72 + 16 + 8 * lh];
      bv1.q[0] = *(const uint4*)&Vt[col * 72 + 32 + 8 * lh];
      bv1.q[1] = *(const uint4*)&Vt[col * 72 + 32 + 16 + 8 * lh];
      o[et] = __builtin_amdgcn_wmma_f32_16x16x32_bf16(
                  false, ap[0].v, false, bv0.v, (short)0, o[et], false, false);
      o[et] = __builtin_amdgcn_wmma_f32_16x16x32_bf16(
                  false, ap[1].v, false, bv1.v, (short)0, o[et], false, false);
    }
  }

  // ---- normalize + store bf16 to [B, L, D] for output projection ----
  const int b = bh >> 4, h = bh & (HEADS - 1);
  #pragma unroll
  for (int r = 0; r < 8; ++r) {
    const float inv = 1.0f / li[r];
    const int qr = q0 + wid * 16 + r + 8 * lh;
    const size_t rowoff = ((size_t)(b * SEQ + qr)) * D_MODEL + h * HDIM;
    #pragma unroll
    for (int et = 0; et < 4; ++et)
      O[rowoff + et * 16 + lm] = f32_bf16(o[et][r] * inv);
  }
}

// ---------------------------------------------------------------------------
extern "C" void kernel_launch(void* const* d_in, const int* in_sizes, int n_in,
                              void* d_out, int out_size, void* d_ws, size_t ws_size,
                              hipStream_t stream) {
  const float* x  = (const float*)d_in[0];
  const float* Wq = (const float*)d_in[1];
  const float* bq = (const float*)d_in[2];
  const float* Wk = (const float*)d_in[3];
  const float* bk = (const float*)d_in[4];
  const float* Wv = (const float*)d_in[5];
  const float* bv = (const float*)d_in[6];
  const float* Wo = (const float*)d_in[7];
  const float* bo = (const float*)d_in[8];

  const size_t elems = (size_t)MROWS * D_MODEL;
  unsigned short* Qb = (unsigned short*)d_ws;    // [B,H,L,E] bf16
  unsigned short* Kb = Qb + elems;
  unsigned short* Vb = Kb + elems;
  unsigned short* Ob = Vb + elems;               // [B,L,D] bf16

  const dim3 g(D_MODEL / 64, MROWS / 128);       // (16, 64)
  gemm_bias_k<0, float><<<g, 256, 0, stream>>>(x, Wq, bq, Qb);
  gemm_bias_k<0, float><<<g, 256, 0, stream>>>(x, Wk, bk, Kb);
  gemm_bias_k<0, float><<<g, 256, 0, stream>>>(x, Wv, bv, Vb);

  attn_k<<<dim3(SEQ / 128, BATCH * HEADS), 256, 0, stream>>>(Qb, Kb, Vb, Ob);

  gemm_bias_k<1, unsigned short><<<g, 256, 0, stream>>>(Ob, Wo, bo, d_out);
}
